// CustomDeepseekDBOModel_28200755265616
// MI455X (gfx1250) — matrix operand order repaired and checked by reading, not compiled, not run on hardware
//
#include <hip/hip_runtime.h>

// ---------------------------------------------------------------------------
// DeepSeek MoE layer for gfx1250 (MI455X): router + dispatch + expert SwiGLU
// (f16 WMMA, f32 accumulate) + shared expert + atomic combine.
// Round 2: double-buffered LDS + software pipelining; f16 A tiles staged via
// GLOBAL_LOAD_ASYNC_TO_LDS_B128 (ASYNCcnt) on the GEMM2 paths.
// ---------------------------------------------------------------------------

typedef _Float16 half8  __attribute__((ext_vector_type(8)));
typedef _Float16 v16h   __attribute__((ext_vector_type(16)));
typedef float    v8f    __attribute__((ext_vector_type(8)));

constexpr int   T_   = 1024;
constexpr int   H_   = 2048;
constexpr int   E_   = 32;
constexpr int   TK_  = 6;
constexpr int   IM_  = 768;    // moe_intermediate_size
constexpr int   ISH_ = 1536;   // shared intermediate
constexpr int   CAP_ = 768;    // per-expert capacity
constexpr float SCALE_ = 16.0f;

constexpr int BM = 128, BN = 64, BK = 32, BKP = BK + 8; // LDS pad breaks conflicts
constexpr int NTHREADS = 256;

// ---------------------------------------------------------------------------

__device__ __forceinline__ half8 cvt8r(float4 a, float4 b) {
    half8 h;
    h[0]=(_Float16)a.x; h[1]=(_Float16)a.y; h[2]=(_Float16)a.z; h[3]=(_Float16)a.w;
    h[4]=(_Float16)b.x; h[5]=(_Float16)b.y; h[6]=(_Float16)b.z; h[7]=(_Float16)b.w;
    return h;
}

__device__ __forceinline__ half8 zero8() {
    half8 h;
#pragma unroll
    for (int i = 0; i < 8; ++i) h[i] = (_Float16)0.0f;
    return h;
}

__device__ __forceinline__ v16h mk16(half8 lo, half8 hi) {
    v16h r;
#pragma unroll
    for (int i = 0; i < 8; ++i) { r[i] = lo[i]; r[i + 8] = hi[i]; }
    return r;
}

__device__ __forceinline__ void atomic_add_f32(float* addr, float v) {
    // native global fp32 atomic add (non-returning) — avoids CAS expansion
    asm volatile("global_atomic_add_f32 %0, %1, off"
                 :
                 : "v"((unsigned long long)(uintptr_t)addr), "v"(v)
                 : "memory");
}

// async copy 32B (2 x b128) global f16 -> LDS; tracked by ASYNCcnt
__device__ __forceinline__ void async_g2l_32B(unsigned lds_addr, const void* gaddr) {
    asm volatile("global_load_async_to_lds_b128 %0, %1, off"
                 :
                 : "v"(lds_addr), "v"((unsigned long long)(uintptr_t)gaddr)
                 : "memory");
    asm volatile("global_load_async_to_lds_b128 %0, %1, off offset:16"
                 :
                 : "v"(lds_addr), "v"((unsigned long long)(uintptr_t)gaddr)
                 : "memory");
}

__device__ __forceinline__ void wait_asynccnt0() {
    asm volatile("s_wait_asynccnt 0x0" ::: "memory");
}

__device__ __forceinline__ float silu(float x) {
    return x / (1.0f + __expf(-x));
}

// ---------------------------------------------------------------------------
// Kernel 1: router — logits, softmax, group-limited top-k (one wave / token)
// ---------------------------------------------------------------------------
__global__ __launch_bounds__(32) void gate_kernel(
    const float* __restrict__ x, const float* __restrict__ gw,
    int* __restrict__ topk_ids, float* __restrict__ topk_w)
{
    const int t = blockIdx.x;
    const int e = threadIdx.x;           // one expert per lane (E_ == 32)

    const float4* xr = (const float4*)(x  + (size_t)t * H_);
    const float4* wr = (const float4*)(gw + (size_t)e * H_);
    float s = 0.0f;
    for (int i = 0; i < H_ / 4; ++i) {
        float4 a = xr[i], b = wr[i];
        s += a.x * b.x + a.y * b.y + a.z * b.z + a.w * b.w;
    }
    // softmax across the wave
    float m = s;
#pragma unroll
    for (int off = 16; off > 0; off >>= 1) m = fmaxf(m, __shfl_xor(m, off, 32));
    float p = __expf(s - m);
    float sum = p;
#pragma unroll
    for (int off = 16; off > 0; off >>= 1) sum += __shfl_xor(sum, off, 32);
    float score = p / sum;

    __shared__ float sc[E_];
    sc[e] = score;
    __syncthreads();

    if (e == 0) {
        // group scores = max over 4 consecutive experts
        float gsc[8]; bool gsel[8];
#pragma unroll
        for (int g = 0; g < 8; ++g) {
            float mg = sc[4 * g];
#pragma unroll
            for (int j = 1; j < 4; ++j) mg = fmaxf(mg, sc[4 * g + j]);
            gsc[g] = mg; gsel[g] = false;
        }
        for (int it = 0; it < 3; ++it) {           // top-3 groups (ties -> low idx)
            int bi = -1; float bv = -1e30f;
            for (int g = 0; g < 8; ++g)
                if (!gsel[g] && gsc[g] > bv) { bv = gsc[g]; bi = g; }
            gsel[bi] = true;
        }
        float ms[E_];
        for (int i = 0; i < E_; ++i) ms[i] = gsel[i >> 2] ? sc[i] : 0.0f;
        for (int it = 0; it < TK_; ++it) {         // top-6 experts
            int bi = 0; float bv = ms[0];
            for (int i = 1; i < E_; ++i)
                if (ms[i] > bv) { bv = ms[i]; bi = i; }
            topk_ids[t * TK_ + it] = bi;
            topk_w [t * TK_ + it] = bv;
            ms[bi] = -1.0f;
        }
    }
}

// ---------------------------------------------------------------------------
// Kernel 2: dispatch — per-expert compacted row lists (lane e owns expert e)
// ---------------------------------------------------------------------------
__global__ __launch_bounds__(32) void dispatch_kernel(
    const int* __restrict__ ids, int* __restrict__ rowmap, int* __restrict__ counts)
{
    const int e = threadIdx.x;
    int c = 0;
    for (int i = 0; i < T_ * TK_; ++i) {
        int id = ids[i];
        if (id == e && c < CAP_) { rowmap[e * CAP_ + c] = i; ++c; }
    }
    counts[e] = c;
}

// ---------------------------------------------------------------------------
// Tiled f16-WMMA GEMM (C = A @ B^T, f32 accumulate), double-buffered LDS,
// software pipelined (fetch k+1 overlaps WMMA on k).
//   MODE 0: routed GEMM1  (gathered x rows, dual-B g/u, SwiGLU -> f16 act)
//   MODE 1: shared GEMM1  (x rows,          dual-B g/u, SwiGLU -> f16 act)
//   MODE 2: routed GEMM2  (f16 act via async-to-LDS, out += SCALE*w*y atomics)
//   MODE 3: shared GEMM2  (f16 act via async-to-LDS, out = y plain store)
// Block tile 128x64, 8 waves (4x2), wave tile 32x32 (2x2 of 16x16), K-step 32.
// ---------------------------------------------------------------------------
template <int MODE>
__global__ __launch_bounds__(NTHREADS) void moe_gemm(
    const float*    __restrict__ Af32,   // x                (modes 0,1)
    const _Float16* __restrict__ Af16,   // activations      (modes 2,3)
    const float*    __restrict__ Bw,     // weights
    _Float16*       __restrict__ actOut, // f16 activations  (modes 0,1)
    float*          __restrict__ out,    // output           (modes 2,3)
    const int*      __restrict__ rowmap,
    const int*      __restrict__ counts,
    const float*    __restrict__ topkw)
{
    constexpr bool DUAL = (MODE <= 1);
    constexpr int  NB   = DUAL ? 2 : 1;
    constexpr int  Kdim = (MODE <= 1) ? H_ : (MODE == 2 ? IM_ : ISH_);
    constexpr int  NSTEPS = Kdim / BK;

    const int tid   = threadIdx.x;
    const int lane  = tid & 31;
    const int wave  = tid >> 5;
    const int waveM = wave & 3;     // 4 M-groups of 32 rows
    const int waveN = wave >> 2;    // 2 N-groups of 32 cols

    const int e     = (MODE == 0 || MODE == 2) ? blockIdx.z : 0;
    const int mBase = blockIdx.y * BM;
    const int nBase = blockIdx.x * BN;
    const int rows  = (MODE == 0 || MODE == 2) ? counts[e] : T_;
    if (mBase >= rows) return;

    // ---- B matrix pointers -------------------------------------------------
    const float* Bg; const float* Bu = nullptr;
    if constexpr (MODE == 0) {
        Bg = Bw + (size_t)e * (2 * IM_) * H_;
        Bu = Bg + (size_t)IM_ * H_;
    } else if constexpr (MODE == 1) {
        Bg = Bw;
        Bu = Bw + (size_t)ISH_ * H_;
    } else if constexpr (MODE == 2) {
        Bg = Bw + (size_t)e * H_ * IM_;
    } else {
        Bg = Bw;
    }

    // ---- per-thread A source row ------------------------------------------
    const int ar  = mBase + (tid >> 1);         // 128 rows x 2 threads/row
    const int akk = (tid & 1) * 16;             // A: 16 halves (32B) per thread
    const float*    aSrc  = nullptr;
    const _Float16* aSrcH = nullptr;
    if constexpr (MODE == 0) {
        if (ar < rows) aSrc = Af32 + (size_t)(rowmap[e * CAP_ + ar] / TK_) * H_;
    } else if constexpr (MODE == 1) {
        aSrc = Af32 + (size_t)ar * H_;
    } else if constexpr (MODE == 2) {
        if (ar < rows) aSrcH = Af16 + ((size_t)e * CAP_ + ar) * IM_;
    } else {
        aSrcH = Af16 + (size_t)ar * ISH_;
    }

    const int br  = tid >> 2;                   // B: 64 rows x 4 threads/row
    const int bkk = (tid & 3) * 8;              // B: 8 floats per thread

    __shared__ __align__(16) _Float16 As[2][BM * BKP];
    __shared__ __align__(16) _Float16 Bs[2][NB * BN * BKP];

    // LDS byte addresses of this thread's A-tile slice (for async copies)
    const unsigned ldsA[2] = {
        (unsigned)(uintptr_t)&As[0][(tid >> 1) * BKP + akk],
        (unsigned)(uintptr_t)&As[1][(tid >> 1) * BKP + akk]
    };

    v8f acc[NB][2][2];
#pragma unroll
    for (int b = 0; b < NB; ++b)
#pragma unroll
        for (int mi = 0; mi < 2; ++mi)
#pragma unroll
            for (int ni = 0; ni < 2; ++ni)
#pragma unroll
                for (int j = 0; j < 8; ++j) acc[b][mi][ni][j] = 0.0f;

    float4 aReg[4];
    float4 bReg[NB][2];

    // issue global fetches for stage s (A goes async to LDS on f16 paths)
    auto fetchStage = [&](int s) {
        const int k0 = s * BK;
        if constexpr (MODE <= 1) {
            if (aSrc) {
                const float4* pa = (const float4*)(aSrc + k0 + akk);
                aReg[0] = pa[0]; aReg[1] = pa[1]; aReg[2] = pa[2]; aReg[3] = pa[3];
            }
        } else {
            if (aSrcH) async_g2l_32B(ldsA[s & 1], aSrcH + k0 + akk);
        }
        const float* pg = Bg + (size_t)(nBase + br) * Kdim + k0 + bkk;
        bReg[0][0] = ((const float4*)pg)[0];
        bReg[0][1] = ((const float4*)pg)[1];
        if constexpr (DUAL) {
            const float* pu = Bu + (size_t)(nBase + br) * Kdim + k0 + bkk;
            bReg[1][0] = ((const float4*)pu)[0];
            bReg[1][1] = ((const float4*)pu)[1];
        }
    };

    // convert staged registers and store into LDS stage buffer
    auto storeStage = [&](int s) {
        const int buf = s & 1;
        if constexpr (MODE <= 1) {
            _Float16* dst = &As[buf][(tid >> 1) * BKP + akk];
            if (aSrc) {
                *(half8*)dst       = cvt8r(aReg[0], aReg[1]);
                *(half8*)(dst + 8) = cvt8r(aReg[2], aReg[3]);
            } else {
                *(half8*)dst       = zero8();
                *(half8*)(dst + 8) = zero8();
            }
        } else {
            if (!aSrcH) {   // invalid rows: zero-fill (async path skipped them)
                _Float16* dst = &As[buf][(tid >> 1) * BKP + akk];
                *(half8*)dst       = zero8();
                *(half8*)(dst + 8) = zero8();
            }
        }
        *(half8*)&Bs[buf][br * BKP + bkk] = cvt8r(bReg[0][0], bReg[0][1]);
        if constexpr (DUAL)
            *(half8*)&Bs[buf][BN * BKP + br * BKP + bkk] = cvt8r(bReg[1][0], bReg[1][1]);
    };

    // ---- pipeline prologue -------------------------------------------------
    fetchStage(0);
    storeStage(0);

    // ---- main loop: 1 barrier / step, fetch(k+1) overlaps wmma(k) ----------
    for (int s = 0; s < NSTEPS; ++s) {
        const int buf = s & 1;
        if constexpr (MODE >= 2) wait_asynccnt0();   // async A stores visible
        __syncthreads();                             // stage `buf` ready for all

        if (s + 1 < NSTEPS) fetchStage(s + 1);       // loads in flight under wmma

        const _Float16* Ab = As[buf];
        const _Float16* Bb = Bs[buf];

        v16h af[2];
#pragma unroll
        for (int mi = 0; mi < 2; ++mi) {
            const int r  = waveM * 32 + mi * 16 + (lane & 15);
            const int ko = (lane < 16) ? 0 : 8;   // A: K {0-7,16-23}/{8-15,24-31}
            half8 lo = *(const half8*)&Ab[r * BKP + ko];
            half8 hi = *(const half8*)&Ab[r * BKP + ko + 16];
            af[mi] = mk16(lo, hi);
        }
#pragma unroll
        for (int b = 0; b < NB; ++b) {
#pragma unroll
            for (int ni = 0; ni < 2; ++ni) {
                const int rn = waveN * 32 + ni * 16 + (lane & 15);
                const int ko = (lane < 16) ? 0 : 8;  // B: K {0-7,16-23}/{8-15,24-31}
                const _Float16* bp = &Bb[b * BN * BKP + rn * BKP + ko];
                half8 lo = *(const half8*)bp;
                half8 hi = *(const half8*)(bp + 16);
                v16h bf = mk16(lo, hi);
#pragma unroll
                for (int mi = 0; mi < 2; ++mi) {
                    acc[b][mi][ni] = __builtin_amdgcn_wmma_f32_16x16x32_f16(
                        false, af[mi], false, bf, (short)0, acc[b][mi][ni],
                        false, false);
                }
            }
        }

        if (s + 1 < NSTEPS) storeStage(s + 1);       // waits loads, cvt, ds_store
    }

    // ---- epilogue ---------------------------------------------------------
    const int lm = (lane < 16) ? 0 : 8;   // C layout: VGPR j -> M=j(+8)
    const int ln = lane & 15;             //           N = lane % 16

#pragma unroll
    for (int mi = 0; mi < 2; ++mi) {
#pragma unroll
        for (int ni = 0; ni < 2; ++ni) {
#pragma unroll
            for (int j = 0; j < 8; ++j) {
                const int row = mBase + waveM * 32 + mi * 16 + lm + j;
                const int col = nBase + waveN * 32 + ni * 16 + ln;
                if constexpr (MODE <= 1) {
                    float g = acc[0][mi][ni][j];
                    float u = acc[1][mi][ni][j];
                    float a = silu(g) * u;
                    if (row < rows) {
                        const int ldAct = (MODE == 0) ? IM_ : ISH_;
                        _Float16* ab = (MODE == 0)
                            ? actOut + (size_t)e * CAP_ * IM_ : actOut;
                        ab[(size_t)row * ldAct + col] = (_Float16)a;
                    }
                } else if constexpr (MODE == 2) {
                    if (row < rows) {
                        const int   flat = rowmap[e * CAP_ + row];
                        const int   t    = flat / TK_;
                        const float w    = topkw[flat];
                        atomic_add_f32(out + (size_t)t * H_ + col,
                                       SCALE_ * w * acc[0][mi][ni][j]);
                    }
                } else {
                    out[(size_t)row * H_ + col] = acc[0][mi][ni][j];
                }
            }
        }
    }
}

// ---------------------------------------------------------------------------
extern "C" void kernel_launch(void* const* d_in, const int* in_sizes, int n_in,
                              void* d_out, int out_size, void* d_ws, size_t ws_size,
                              hipStream_t stream)
{
    (void)in_sizes; (void)n_in; (void)out_size; (void)ws_size;

    const float* x    = (const float*)d_in[0];   // [T,H]
    const float* gw   = (const float*)d_in[1];   // [E,H]
    const float* w13  = (const float*)d_in[2];   // [E,2I,H]
    const float* w2   = (const float*)d_in[3];   // [E,H,I]
    const float* sgu  = (const float*)d_in[4];   // [2*ISH,H]
    const float* sdn  = (const float*)d_in[5];   // [H,ISH]
    float*       out  = (float*)d_out;           // [T,H]

    // workspace carve-out (≈41 MB)
    char* w = (char*)d_ws;
    int*       topk_ids = (int*)w;        w += (size_t)T_ * TK_ * sizeof(int);
    float*     topk_w   = (float*)w;      w += (size_t)T_ * TK_ * sizeof(float);
    int*       counts   = (int*)w;        w += 256;
    int*       rowmap   = (int*)w;        w += (size_t)E_ * CAP_ * sizeof(int);
    _Float16*  act_r    = (_Float16*)w;   w += (size_t)E_ * CAP_ * IM_ * sizeof(_Float16);
    _Float16*  act_sh   = (_Float16*)w;   w += (size_t)T_ * ISH_ * sizeof(_Float16);

    // 1) router
    gate_kernel<<<dim3(T_), dim3(32), 0, stream>>>(x, gw, topk_ids, topk_w);
    // 2) dispatch
    dispatch_kernel<<<dim3(1), dim3(32), 0, stream>>>(topk_ids, rowmap, counts);
    // 3) routed GEMM1 + SwiGLU -> f16 act
    moe_gemm<0><<<dim3(IM_ / BN, CAP_ / BM, E_), NTHREADS, 0, stream>>>(
        x, nullptr, w13, act_r, nullptr, rowmap, counts, topk_w);
    // 4) shared GEMM1 + SwiGLU -> f16 act
    moe_gemm<1><<<dim3(ISH_ / BN, T_ / BM, 1), NTHREADS, 0, stream>>>(
        x, nullptr, sgu, act_sh, nullptr, rowmap, counts, topk_w);
    // 5) shared GEMM2 -> out (plain store, runs BEFORE routed atomics)
    moe_gemm<3><<<dim3(H_ / BN, T_ / BM, 1), NTHREADS, 0, stream>>>(
        nullptr, act_sh, sdn, nullptr, out, rowmap, counts, topk_w);
    // 6) routed GEMM2 -> out += SCALE * w * y (native f32 atomics)
    moe_gemm<2><<<dim3(H_ / BN, CAP_ / BM, E_), NTHREADS, 0, stream>>>(
        nullptr, act_r, w2, nullptr, out, rowmap, counts, topk_w);
}
